// GNN_89799176224972
// MI455X (gfx1250) — compile-verified, matrix operand order checked by reading, not run on hardware
//
#include <hip/hip_runtime.h>
#include <hip/hip_bf16.h>

#define NN 50000
#define NE 800000

typedef __attribute__((ext_vector_type(16))) _Float16 v16h;
typedef __attribute__((ext_vector_type(8)))  _Float16 half8;
typedef __attribute__((ext_vector_type(8)))  float    v8f;

// Load a v16h A/B fragment as two contiguous 8-half (16B) runs.
__device__ __forceinline__ v16h ld16(const _Float16* p0, const _Float16* p1) {
    half8 a = *(const half8*)p0;
    half8 b = *(const half8*)p1;
    return __builtin_shufflevector(a, b, 0,1,2,3,4,5,6,7,8,9,10,11,12,13,14,15);
}

__device__ __forceinline__ v8f wmma_f16(v16h a, v16h b, v8f c) {
    return __builtin_amdgcn_wmma_f32_16x16x32_f16(false, a, false, b, (short)0, c, false, false);
}

__global__ void k_zero(float* p, int n) {
    int i = blockIdx.x * blockDim.x + threadIdx.x;
    if (i < n) p[i] = 0.0f;
}

__global__ void k_count(const int* __restrict__ dst, float* __restrict__ cnt, int nE) {
    int i = blockIdx.x * blockDim.x + threadIdx.x;
    if (i < nE) atomicAdd(&cnt[dst[i]], 1.0f);
}

// Fused edge layer: msg = relu(concat(feat[src],feat[dst]) @ We + be);
// agg[dst] += msg.   Block = 256 threads (8 waves), 128 edges per block.
__global__ __launch_bounds__(256) void k_edge(const float* __restrict__ feat,
                                              const int*   __restrict__ src,
                                              const int*   __restrict__ dst,
                                              const float* __restrict__ We,   // [128,64] row-major
                                              const float* __restrict__ be,   // [64]
                                              float*       __restrict__ agg,  // [NN,64]
                                              int nE) {
    __shared__ alignas(16) _Float16 sE[128 * 136];  // edge features (f16), padded stride
    __shared__ alignas(16) _Float16 sW[64 * 136];   // Wt[n][k] (f16), padded stride
    __shared__ int   sDst[128];
    __shared__ float sB[64];

    const int tid = threadIdx.x;
    const int eb  = blockIdx.x * 128;

    // Stage transposed weights: sW[n*136+k] = We[k*64+n]  (i = k*64+n, coalesced read)
    for (int i = tid; i < 64 * 128; i += 256) {
        int k = i >> 6, n = i & 63;
        sW[n * 136 + k] = (_Float16)We[i];
    }
    if (tid < 64)  sB[tid] = be[tid];
    if (tid < 128) {
        int ge = eb + tid; if (ge >= nE) ge = nE - 1;
        sDst[tid] = dst[ge];
    }

    // Gather 128 edges x 128 f16 features (src 0..63 | dst 64..127), 16 feats/chunk
    #pragma unroll
    for (int it = 0; it < 4; ++it) {
        int idx = it * 256 + tid;           // 0..1023
        int e = idx >> 3, c = idx & 7;
        int ge = eb + e; if (ge >= nE) ge = nE - 1;
        int node = (c < 4) ? src[ge] : dst[ge];
        const float4* gp = (const float4*)(feat + (long)node * 64 + (c & 3) * 16);
        float4 x0 = gp[0], x1 = gp[1], x2 = gp[2], x3 = gp[3];
        half8 h0 = { (_Float16)x0.x, (_Float16)x0.y, (_Float16)x0.z, (_Float16)x0.w,
                     (_Float16)x1.x, (_Float16)x1.y, (_Float16)x1.z, (_Float16)x1.w };
        half8 h1 = { (_Float16)x2.x, (_Float16)x2.y, (_Float16)x2.z, (_Float16)x2.w,
                     (_Float16)x3.x, (_Float16)x3.y, (_Float16)x3.z, (_Float16)x3.w };
        _Float16* lp = &sE[e * 136 + c * 16];
        *(half8*)lp       = h0;
        *(half8*)(lp + 8) = h1;
    }
    __syncthreads();

    // Each wave: 16 edges x 64 outputs = 4 N-tiles, K=128 in 4 steps of 32.
    const int w = tid >> 5, lane = tid & 31;
    const int lr = lane & 15, hi = lane >> 4;
    const _Float16* erow = &sE[(w * 16 + lr) * 136];

    v8f acc[4] = {};
    #pragma unroll
    for (int kb = 0; kb < 128; kb += 32) {
        v16h a = ld16(erow + kb + hi * 8, erow + kb + 16 + hi * 8);
        #pragma unroll
        for (int nt = 0; nt < 4; ++nt) {
            const _Float16* wrow = &sW[(nt * 16 + lr) * 136];
            v16h b = ld16(wrow + kb + hi * 8, wrow + kb + 16 + hi * 8);
            acc[nt] = wmma_f16(a, b, acc[nt]);
        }
    }

    // bias + relu + scatter-add.  D layout: vgpr r, lane -> edge r+hi*8, col lane%16
    #pragma unroll
    for (int nt = 0; nt < 4; ++nt) {
        float bia = sB[nt * 16 + lr];
        #pragma unroll
        for (int r = 0; r < 8; ++r) {
            int el = w * 16 + r + hi * 8;
            int ge = eb + el;
            float v = acc[nt][r] + bia;
            v = v > 0.0f ? v : 0.0f;
            if (ge < nE)
                atomicAdd(&agg[(long)sDst[el] * 64 + nt * 16 + lr], v);
        }
    }
}

// Node update: out = relu((agg / max(cnt,1)) @ Wn + bn).  128 nodes per block.
__global__ __launch_bounds__(256) void k_node(const float* __restrict__ agg,
                                              const float* __restrict__ cnt,
                                              const float* __restrict__ Wn,   // [64,64]
                                              const float* __restrict__ bn,
                                              float*       __restrict__ out,  // [NN,64]
                                              int nN) {
    __shared__ alignas(16) _Float16 sW[64 * 72];
    __shared__ float sB[64];

    const int tid = threadIdx.x;
    for (int i = tid; i < 64 * 64; i += 256) {
        int k = i >> 6, n = i & 63;
        sW[n * 72 + k] = (_Float16)Wn[i];
    }
    if (tid < 64) sB[tid] = bn[tid];
    __syncthreads();

    const int w = tid >> 5, lane = tid & 31;
    const int lr = lane & 15, hi = lane >> 4;
    const int nBase = blockIdx.x * 128 + w * 16;

    int row = nBase + lr; if (row >= nN) row = nN - 1;
    float rinv = 1.0f / fmaxf(cnt[row], 1.0f);
    const float* ar = agg + (long)row * 64;

    v8f acc[4] = {};
    #pragma unroll
    for (int kb = 0; kb < 64; kb += 32) {
        float4 x0 = *(const float4*)(ar + kb + hi * 8);
        float4 x1 = *(const float4*)(ar + kb + hi * 8 + 4);
        float4 x2 = *(const float4*)(ar + kb + 16 + hi * 8);
        float4 x3 = *(const float4*)(ar + kb + 16 + hi * 8 + 4);
        v16h a = { (_Float16)(x0.x*rinv), (_Float16)(x0.y*rinv), (_Float16)(x0.z*rinv), (_Float16)(x0.w*rinv),
                   (_Float16)(x1.x*rinv), (_Float16)(x1.y*rinv), (_Float16)(x1.z*rinv), (_Float16)(x1.w*rinv),
                   (_Float16)(x2.x*rinv), (_Float16)(x2.y*rinv), (_Float16)(x2.z*rinv), (_Float16)(x2.w*rinv),
                   (_Float16)(x3.x*rinv), (_Float16)(x3.y*rinv), (_Float16)(x3.z*rinv), (_Float16)(x3.w*rinv) };
        #pragma unroll
        for (int nt = 0; nt < 4; ++nt) {
            const _Float16* wrow = &sW[(nt * 16 + lr) * 72];
            v16h b = ld16(wrow + kb + hi * 8, wrow + kb + 16 + hi * 8);
            acc[nt] = wmma_f16(a, b, acc[nt]);
        }
    }

    #pragma unroll
    for (int nt = 0; nt < 4; ++nt) {
        float bia = sB[nt * 16 + lr];
        #pragma unroll
        for (int r = 0; r < 8; ++r) {
            int node = nBase + r + hi * 8;
            if (node < nN) {
                float v = acc[nt][r] + bia;
                out[(long)node * 64 + nt * 16 + lr] = fmaxf(v, 0.0f);
            }
        }
    }
}

extern "C" void kernel_launch(void* const* d_in, const int* in_sizes, int n_in,
                              void* d_out, int out_size, void* d_ws, size_t ws_size,
                              hipStream_t stream) {
    const float* nf  = (const float*)d_in[0];
    const int*   src = (const int*)  d_in[1];
    const int*   dst = (const int*)  d_in[2];
    const float* W0e = (const float*)d_in[3];
    const float* b0e = (const float*)d_in[4];
    const float* W0n = (const float*)d_in[5];
    const float* b0n = (const float*)d_in[6];
    const float* W1e = (const float*)d_in[7];
    const float* b1e = (const float*)d_in[8];
    const float* W1n = (const float*)d_in[9];
    const float* b1n = (const float*)d_in[10];
    float* out = (float*)d_out;

    float* agg = (float*)d_ws;                 // [NN*64]
    float* cnt = agg + (size_t)NN * 64;        // [NN]

    const int totz = NN * 64 + NN;
    k_zero <<<(totz + 255) / 256, 256, 0, stream>>>(agg, totz);
    k_count<<<(NE + 255) / 256,  256, 0, stream>>>(dst, cnt, NE);

    // Layer 0
    k_edge <<<(NE + 127) / 128, 256, 0, stream>>>(nf, src, dst, W0e, b0e, agg, NE);
    k_node <<<(NN + 127) / 128, 256, 0, stream>>>(agg, cnt, W0n, b0n, out, NN);

    // Layer 1 (hidden state lives in d_out)
    k_zero <<<(NN * 64 + 255) / 256, 256, 0, stream>>>(agg, NN * 64);
    k_edge <<<(NE + 127) / 128, 256, 0, stream>>>(out, src, dst, W1e, b1e, agg, NE);
    k_node <<<(NN + 127) / 128, 256, 0, stream>>>(agg, cnt, W1n, b1n, out, NN);
}
